// WinSelfAttentionOperator_19000935317649
// MI455X (gfx1250) — matrix operand order repaired and checked by reading, not compiled, not run on hardware
//
#include <hip/hip_runtime.h>

// ---------------- constants ----------------
constexpr int kB  = 2;     // batch
constexpr int kS  = 64;    // tokens per window
constexpr int kC  = 64;    // per-head channels
constexpr int kNH = 8;     // heads
constexpr int kH  = 32;
constexpr int kW  = 32;
constexpr int kM  = 16;    // retained modes per axis
constexpr int kNM = 256;   // kM*kM
constexpr int kCO = 512;   // kNH*kC
constexpr float kPI = 3.14159265358979323846f;
// scale/(H*W) = (1/256)/1024
constexpr float kBase = 3.814697265625e-06f;

typedef __attribute__((ext_vector_type(16))) __bf16 v16bf;
typedef __attribute__((ext_vector_type(8)))  float  v8f;
typedef unsigned int v4u __attribute__((ext_vector_type(4)));
typedef int v4i __attribute__((ext_vector_type(4)));
typedef int v8i __attribute__((ext_vector_type(8)));

__device__ __forceinline__ v8f wmma_bf16(v16bf a, v16bf b, v8f c) {
  return __builtin_amdgcn_wmma_f32_16x16x32_bf16(false, a, false, b, (short)0, c, false, false);
}
__device__ __forceinline__ v8f v8f_zero() {
  v8f z = {0.f,0.f,0.f,0.f,0.f,0.f,0.f,0.f};
  return z;
}

// ---- Tensor Data Mover: 1-D contiguous DMA global->LDS (D# per ISA 08_async_tensor §8) ----
__device__ __forceinline__ void tdm_load_1d(unsigned lds_byte_off, const void* gsrc,
                                            unsigned ndwords) {
  unsigned long long ga = (unsigned long long)(size_t)gsrc;
  v4u g0;
  g0[0] = 1u;                                         // count=1, user descriptor
  g0[1] = lds_byte_off;                               // lds_addr [63:32]
  g0[2] = (unsigned)(ga & 0xffffffffu);               // global_addr[31:0]
  g0[3] = (unsigned)((ga >> 32) & 0x01ffffffu)        // global_addr[56:32]
          | 0x80000000u;                              // type=2 ("image") in [127:126]
  v8i g1;
  g1[0] = (int)(2u << 16);                            // wg_mask=0, data_size=2 (4 bytes)
  g1[1] = (int)((ndwords & 0xffffu) << 16);           // tensor_dim0[15:0] @ [63:48]
  g1[2] = (int)(((ndwords >> 16) & 0xffffu)           // tensor_dim0[31:16] @ [79:64]
          | (1u << 16));                              // tensor_dim1 = 1 @ [111:80]
  g1[3] = (int)((ndwords & 0xffffu) << 16);           // tile_dim0 @ [127:112]
  g1[4] = 1;                                          // tile_dim1 = 1, tile_dim2 = 0
  g1[5] = (int)ndwords;                               // tensor_dim0_stride[31:0]
  g1[6] = 0;
  g1[7] = 0;
  v4i gz = {0, 0, 0, 0};
#if __has_include(<hip/amd_detail/amd_gfx1250_TDM.h>)
  v8i gz8 = {0, 0, 0, 0, 0, 0, 0, 0};
  __builtin_amdgcn_tensor_load_to_lds(g0, g1, gz, gz, gz8, 0);
#else
  __builtin_amdgcn_tensor_load_to_lds(g0, g1, gz, gz, 0);
#endif
}
__device__ __forceinline__ unsigned lds_off32(const void* p) {
  return (unsigned)(size_t)p;                         // LDS pointers are 32-bit offsets
}

// ---------------- weight prep: qkv weights -> bf16 B-native tiles ----------------
// out layout: [mode][chunk(4)][coltile(32)][lane(32)][e(16)]   (k' = cin*2+p, K=128)
__global__ void prep_w_qkv(const float* __restrict__ w, __bf16* __restrict__ out) {
  int tid = blockIdx.x * 256 + threadIdx.x;       // 1,048,576 total
  int lane = tid & 31;
  int coltile = (tid >> 5) & 31;
  int chunk = (tid >> 10) & 3;
  int mode = tid >> 12;
  int kx = mode >> 4, ky = mode & 15;
  v16bf av;
#pragma unroll
  for (int e = 0; e < 16; ++e) {
    int kk = (e < 8) ? ((lane >> 4) * 8 + e) : (16 + (lane >> 4) * 8 + (e - 8));  // B-native: table k order
    int kp = chunk * 32 + kk;                     // k' = cin*2+p
    int cin = kp >> 1, p = kp & 1;
    int cout = coltile * 16 + (lane & 15);
    float v = w[(((size_t)cin * kCO + cout) * kM + kx) * (kM * 2) + ky * 2 + p];
    av[e] = (__bf16)v;
  }
  *(v16bf*)(out + (size_t)tid * 16) = av;
}

// ---------------- weight prep: wo -> two variants (re/im) ----------------
// out layout: [var(2)][mode][chunk(32)][coltile(4)][lane][16]  (k'' = hc*2+p, K=1024)
// var0 (re): [even]=Wr, [odd]=-Wi ; var1 (im): [even]=Wi, [odd]=Wr
__global__ void prep_w_o(const float* __restrict__ w, __bf16* __restrict__ out) {
  int tid = blockIdx.x * 256 + threadIdx.x;       // 2,097,152 total
  int lane = tid & 31;
  int coltile = (tid >> 5) & 3;
  int chunk = (tid >> 7) & 31;
  int mode = (tid >> 12) & 255;
  int var = tid >> 20;
  int kx = mode >> 4, ky = mode & 15;
  v16bf av;
#pragma unroll
  for (int e = 0; e < 16; ++e) {
    int kk = (e < 8) ? ((lane >> 4) * 8 + e) : (16 + (lane >> 4) * 8 + (e - 8));
    int kp = chunk * 32 + kk;                     // hc*2+p
    int hc = kp >> 1, p = kp & 1;
    int o = coltile * 16 + (lane & 15);
    size_t bidx = (((size_t)hc * kC + o) * kM + kx) * (kM * 2) + ky * 2;
    float wr = w[bidx], wi = w[bidx + 1];
    float v = (var == 0) ? (p == 0 ? wr : -wi) : (p == 0 ? wi : wr);
    av[e] = (__bf16)v;
  }
  *(v16bf*)(out + (size_t)tid * 16) = av;
}

// ---------------- continuous position bias ----------------
__device__ __forceinline__ float cpb_norm(float x) {
  float l = log2f(fabsf(x) + 1.0f) * 0.3333333333333333f;  // /log2(8)
  return x < 0.0f ? -l : l;
}
__global__ void cpb_bias_k(const float* __restrict__ w1, const float* __restrict__ b1,
                           const float* __restrict__ w2, const float* __restrict__ b2,
                           float* __restrict__ bias) {
  int id = blockIdx.x * 256 + threadIdx.x;
  if (id >= kS * kS) return;
  int s = id / kS, t = id % kS;
  int sx = s >> 3, sy = s & 7, tx = t >> 3, ty = t & 7;
  float r0 = cpb_norm((float)(sx - tx) * (8.0f / 7.0f));
  float r1 = cpb_norm((float)(sy - ty) * (8.0f / 7.0f));
  float outh[kNH];
#pragma unroll
  for (int h = 0; h < kNH; ++h) outh[h] = b2[h];
  for (int j = 0; j < kC; ++j) {
    float hj = r0 * w1[j] + r1 * w1[kC + j] + b1[j];
    hj = hj > 0.f ? hj : 0.f;
#pragma unroll
    for (int h = 0; h < kNH; ++h) outh[h] += hj * w2[j * kNH + h];
  }
#pragma unroll
  for (int h = 0; h < kNH; ++h)
    bias[((size_t)h * kS + s) * kS + t] = 16.0f / (1.0f + __expf(-outh[h]));
}

// ---------------- forward truncated rfft2 -> A-native bf16, 2 variants ----------------
// xa layout: [var(2)][mode][strip(8)][chunk(4)][lane][16]   rows = bs (128), k' = c*2+p
// var0 = (Xr, -Xi) (conj), var1 = (Xi, Xr) (swap)
__global__ void dft_fwd(const float* __restrict__ seq, __bf16* __restrict__ xa) {
  __shared__ float tile[kH * kW];
  __shared__ float g1r[32 * 16], g1i[32 * 16];
  int blk = blockIdx.x;                   // bs*64 + c
  int bs = blk >> 6, c = blk & 63;
  const float* src = seq + (size_t)blk * (kH * kW);
  int t = threadIdx.x;
  for (int i = t; i < kH * kW; i += 256) tile[i] = src[i];
  __syncthreads();
  for (int id = t; id < 512; id += 256) {           // stage 1: over y
    int x = id >> 4, ky = id & 15;
    float ar = 0.f, ai = 0.f;
    for (int y = 0; y < 32; ++y) {
      float sn, cs;
      __sincosf(-2.0f * kPI * (float)(ky * y) / 32.0f, &sn, &cs);
      float v = tile[x * 32 + y];
      ar += v * cs; ai += v * sn;
    }
    g1r[id] = ar; g1i[id] = ai;
  }
  __syncthreads();
  {                                                  // stage 2: over x, one mode/thread
    int kx = t >> 4, ky = t & 15;
    float ar = 0.f, ai = 0.f;
    for (int x = 0; x < 32; ++x) {
      float sn, cs;
      __sincosf(-2.0f * kPI * (float)(kx * x) / 32.0f, &sn, &cs);
      float gr = g1r[x * 16 + ky], gi = g1i[x * 16 + ky];
      ar += gr * cs - gi * sn;
      ai += gr * sn + gi * cs;
    }
    int mode = t;
    int strip = bs >> 4, r = bs & 15;
    int kp = c * 2;
    int chunk = kp >> 5, kk = kp & 31;
    int lane = r + (((kk & 15) >= 8) ? 16 : 0);
    int e = (kk & 7) + ((kk >= 16) ? 8 : 0);
    size_t v0 = ((((size_t)0 * kNM + mode) * 8 + strip) * 4 + chunk) * 512 + lane * 16 + e;
    size_t v1 = ((((size_t)1 * kNM + mode) * 8 + strip) * 4 + chunk) * 512 + lane * 16 + e;
    xa[v0] = (__bf16)ar;  xa[v0 + 1] = (__bf16)(-ai);
    xa[v1] = (__bf16)ai;  xa[v1 + 1] = (__bf16)ar;
  }
}

// ---------------- per-mode complex projection GEMM [128x64(cplx)] x [64x512(cplx)] ----------------
// kind 0: Q (weighted, -> score A-native), 1: K (-> score B-native), 2: V (-> pass2 B-native)
__global__ void proj_qkv(const __bf16* __restrict__ xa, const __bf16* __restrict__ wb,
                         __bf16* __restrict__ out, int kind) {
  int mode = blockIdx.x >> 3;
  int strip = blockIdx.x & 7;
  int wave = threadIdx.x >> 5, lane = threadIdx.x & 31;
  int ct0 = wave * 4;
  const __bf16* xa0 = xa + ((((size_t)0 * kNM + mode) * 8 + strip) * 4) * 512;
  const __bf16* xa1 = xa + ((((size_t)1 * kNM + mode) * 8 + strip) * 4) * 512;
  v8f accr[4], acci[4];
#pragma unroll
  for (int i = 0; i < 4; ++i) { accr[i] = v8f_zero(); acci[i] = v8f_zero(); }
#pragma unroll
  for (int ch = 0; ch < 4; ++ch) {
    v16bf a0 = *(const v16bf*)(xa0 + (size_t)ch * 512 + lane * 16);
    v16bf a1 = *(const v16bf*)(xa1 + (size_t)ch * 512 + lane * 16);
#pragma unroll
    for (int tt = 0; tt < 4; ++tt) {
      v16bf bm = *(const v16bf*)(wb + (((size_t)mode * 4 + ch) * 32 + (ct0 + tt)) * 512 + lane * 16);
      accr[tt] = wmma_bf16(a0, bm, accr[tt]);
      acci[tt] = wmma_bf16(a1, bm, acci[tt]);
    }
  }
  int kx = mode >> 4, ky = mode & 15;
  float fre = 1.f, fim = 1.f;
  if (kind == 0) {                 // fold Parseval weights + attention scale into Q
    if (ky >= 1)      { fre = 2.f * kBase; fim = 2.f * kBase; }
    else if (kx >= 1) { fre = 0.5f * kBase; fim = 0.5f * kBase; }
    else              { fre = kBase;        fim = 0.f; }
  }
  int n = lane & 15, rlo = 8 * (lane >> 4);
#pragma unroll
  for (int tt = 0; tt < 4; ++tt) {
    int hc = (ct0 + tt) * 16 + n;
    int h = hc >> 6, cc = hc & 63;
#pragma unroll
    for (int j = 0; j < 8; ++j) {
      int bsrow = strip * 16 + j + rlo;
      int b = bsrow >> 6, s = bsrow & 63;
      int sstrip = s >> 4, r = s & 15;
      float vr = accr[tt][j] * fre, vi = acci[tt][j] * fim;
      if (kind == 0) {             // score A-native: [bh][strip4][chunk1024][lane][16]
        int kp0 = (mode * 64 + cc) * 2;
        int chq = kp0 >> 5, kk = kp0 & 31;
        int lane_ = r + (((kk & 15) >= 8) ? 16 : 0);
        int ei = (kk & 7) + ((kk >= 16) ? 8 : 0);
        size_t idx = ((((size_t)(b * kNH + h) * 4 + sstrip) * 1024 + chq) * 32 + lane_) * 16 + ei;
        out[idx] = (__bf16)vr; out[idx + 1] = (__bf16)vi;
      } else if (kind == 1) {      // score B-native: [bh][coltile4][chunk1024][lane][16]
        int kp0 = (mode * 64 + cc) * 2;
        int chq = kp0 >> 5, kk = kp0 & 31;
        int lane_ = r + (((kk & 15) >= 8) ? 16 : 0);
        int ei = (kk & 7) + ((kk >= 16) ? 8 : 0);
        size_t idx = ((((size_t)(b * kNH + h) * 4 + sstrip) * 1024 + chq) * 32 + lane_) * 16 + ei;
        out[idx] = (__bf16)vr; out[idx + 1] = (__bf16)vi;
      } else {                     // V B-native: [b][mode][coltile64][chunk_t2][lane][16]
        int jj = hc * 2;
        int colt = jj >> 4, nn = jj & 15;
        int cht = s >> 5, kkt = s & 31;
        int lane_ = nn + (((kkt & 15) >= 8) ? 16 : 0);
        int ei = (kkt & 7) + ((kkt >= 16) ? 8 : 0);
        size_t idx = ((((size_t)(b * kNM + mode) * 64 + colt) * 2 + cht) * 32 + lane_) * 16 + ei;
        out[idx] = (__bf16)vr; out[idx + 16] = (__bf16)vi;
      }
    }
  }
}

// ---------------- scores: K-split GEMM, TDM double-buffered LDS staging ----------------
// grid = 16 bh * 8 ksegments; each segment covers 128 of 1024 chunks.
// dynamic LDS: A[2][4 strip][8 chunk][512] + B[2][4 coltile][8 chunk][512] bf16 = 128 KB
__global__ void score_gemm_tdm(const __bf16* __restrict__ qs, const __bf16* __restrict__ kb,
                               float* __restrict__ partial) {
  extern __shared__ __bf16 smem[];
  __bf16* sA = smem;                       // [buf][slot][chunk8][512]
  __bf16* sB = smem + 2 * 4 * 8 * 512;
  int bh = blockIdx.x >> 3;
  int seg = blockIdx.x & 7;
  int wave = threadIdx.x >> 5, lane = threadIdx.x & 31;
  int strip = wave >> 1, ct0 = (wave & 1) * 2;
  int ch_base = seg * 128;

  int slot = wave & 3;
  bool isA = wave < 4;
  const __bf16* gsrc = (isA ? qs : kb) + (((size_t)bh * 4 + slot) * 1024 + ch_base) * 512;
  __bf16* lds_slot = (isA ? sA : sB) + slot * (8 * 512);

  // prime buffer 0
  tdm_load_1d(lds_off32(lds_slot), gsrc, 2048);
  v8f acc0 = v8f_zero(), acc1 = v8f_zero();
  for (int t = 0; t < 16; ++t) {                   // 16 tiles x 8 chunks = 128 chunks
    int buf = t & 1;
    if (t + 1 < 16) {                              // prefetch next tile into other buffer
      tdm_load_1d(lds_off32(lds_slot + (1 - buf) * (4 * 8 * 512)),
                  gsrc + (size_t)(t + 1) * 8 * 512, 2048);
      __builtin_amdgcn_s_wait_tensorcnt(1);        // tile t complete (in-order per wave)
    } else {
      __builtin_amdgcn_s_wait_tensorcnt(0);        // drain last tile
    }
    __syncthreads();                               // publish tile t to all waves
    const __bf16* A  = sA + (buf * 4 + strip) * (8 * 512);
    const __bf16* B0 = sB + (buf * 4 + ct0) * (8 * 512);
    const __bf16* B1 = sB + (buf * 4 + ct0 + 1) * (8 * 512);
#pragma unroll
    for (int c = 0; c < 8; ++c) {
      v16bf a  = *(const v16bf*)(A + c * 512 + lane * 16);
      v16bf k0 = *(const v16bf*)(B0 + c * 512 + lane * 16);
      v16bf k1 = *(const v16bf*)(B1 + c * 512 + lane * 16);
      acc0 = wmma_bf16(a, k0, acc0);
      acc1 = wmma_bf16(a, k1, acc1);
    }
    __syncthreads();                               // done reading buf before its reuse
  }
  int n = lane & 15, rlo = 8 * (lane >> 4);
#pragma unroll
  for (int j = 0; j < 8; ++j) {
    int sg = strip * 16 + j + rlo;
    int t0 = ct0 * 16 + n, t1 = t0 + 16;
    partial[(((size_t)seg * 16 + bh) * 64 + sg) * 64 + t0] = acc0[j];
    partial[(((size_t)seg * 16 + bh) * 64 + sg) * 64 + t1] = acc1[j];
  }
}

// ---------------- reduce K-split partials + bias ----------------
__global__ void score_reduce(const float* __restrict__ partial, const float* __restrict__ bias,
                             float* __restrict__ scores) {
  int id = blockIdx.x * 256 + threadIdx.x;         // 65536 = 16*64*64
  if (id >= kB * kNH * kS * kS) return;
  int bh = id >> 12, st = id & 4095;
  int s = st >> 6, t = st & 63;
  float acc = bias[((size_t)(bh & 7) * 64 + s) * 64 + t];
#pragma unroll
  for (int seg = 0; seg < 8; ++seg) acc += partial[((size_t)seg * 16 + bh) * 4096 + st];
  scores[id] = acc;
}

// ---------------- softmax -> attn bf16 in pass-2 A-native layout ----------------
// attn layout: [bh][strip4][chunk2][lane][16]
__global__ void softmax_k(const float* __restrict__ scores, __bf16* __restrict__ attn) {
  int id = blockIdx.x * 256 + threadIdx.x;
  if (id >= kB * kNH * kS) return;
  int bh = id >> 6, s = id & 63;
  const float* row = scores + (size_t)id * 64;
  float mx = row[0];
  for (int t = 1; t < 64; ++t) mx = fmaxf(mx, row[t]);
  float sum = 0.f;
  for (int t = 0; t < 64; ++t) sum += __expf(row[t] - mx);
  float inv = 1.0f / sum;
  int strip = s >> 4, r = s & 15;
  for (int t = 0; t < 64; ++t) {
    float a = __expf(row[t] - mx) * inv;
    int ch = t >> 5, kk = t & 31;
    int lane_ = r + (((kk & 15) >= 8) ? 16 : 0);
    int ei = (kk & 7) + ((kk >= 16) ? 8 : 0);
    size_t idx = ((((size_t)bh * 4 + strip) * 2 + ch) * 32 + lane_) * 16 + ei;
    attn[idx] = (__bf16)a;
  }
}

// ---------------- pass 2: Ohat = attn x Vhat per (b, mode, strip) ----------------
// oa layout (A-native for out_proj): [b][mode][strip4][chunk32][lane][16]
// folds the re-rfft2 ky==0 correction (x0.5 / DC-real) for the output projection
__global__ void attn_v(const __bf16* __restrict__ attn, const __bf16* __restrict__ vb,
                       __bf16* __restrict__ oa) {
  int blk = blockIdx.x;
  int strip = blk & 3;
  int tm = blk >> 2;
  int mode = tm & 255, b = tm >> 8;
  int wave = threadIdx.x >> 5, lane = threadIdx.x & 31;
  int bh = b * kNH + wave;                       // wave's 8 coltiles all map to head = wave
  v8f acc[8];
#pragma unroll
  for (int u = 0; u < 8; ++u) acc[u] = v8f_zero();
  const __bf16* abase = attn + (((size_t)bh * 4 + strip) * 2) * 512;
  const __bf16* vbase = vb + (((size_t)(b * kNM + mode) * 64 + wave * 8) * 2) * 512;
#pragma unroll
  for (int ch = 0; ch < 2; ++ch) {
    v16bf a = *(const v16bf*)(abase + (size_t)ch * 512 + lane * 16);
#pragma unroll
    for (int u = 0; u < 8; ++u) {
      v16bf bm = *(const v16bf*)(vbase + ((size_t)u * 2 + ch) * 512 + lane * 16);
      acc[u] = wmma_bf16(a, bm, acc[u]);
    }
  }
  int kx = mode >> 4, ky = mode & 15;
  float fre = (ky >= 1) ? 1.f : ((kx >= 1) ? 0.5f : 1.f);
  float fim = (ky >= 1) ? 1.f : ((kx >= 1) ? 0.5f : 0.f);
  int n = lane & 15, rlo = 8 * (lane >> 4);
#pragma unroll
  for (int u = 0; u < 8; ++u) {
    int jt = wave * 8 + u;
#pragma unroll
    for (int j = 0; j < 8; ++j) {
      int jcol = jt * 16 + n;                    // 0..1023 = hc*2+p
      int p = jcol & 1;
      float v = acc[u][j] * (p ? fim : fre);
      int chq = jcol >> 5, kk = jcol & 31;
      int lane_ = (j + rlo) + (((kk & 15) >= 8) ? 16 : 0);
      int ei = (kk & 7) + ((kk >= 16) ? 8 : 0);
      size_t idx = ((((size_t)(b * kNM + mode) * 4 + strip) * 32 + chq) * 32 + lane_) * 16 + ei;
      oa[idx] = (__bf16)v;
    }
  }
}

// ---------------- output projection: Yhat = Ohat x Wo per (b, mode) ----------------
// yh layout: [b][s][o][mode][2] f32  (iDFT-friendly)
__global__ void out_proj(const __bf16* __restrict__ oa, const __bf16* __restrict__ wob,
                         float* __restrict__ yh) {
  int blk = blockIdx.x;                          // b*kNM + mode
  int mode = blk & 255, b = blk >> 8;
  int wave = threadIdx.x >> 5, lane = threadIdx.x & 31;
  int strip = wave >> 1, ct0 = (wave & 1) * 2;
  const __bf16* abase = oa + (((size_t)blk * 4 + strip) * 32) * 512;
  const __bf16* wr = wob + (((size_t)0 * kNM + mode) * 32 * 4) * 512;
  const __bf16* wi = wob + (((size_t)1 * kNM + mode) * 32 * 4) * 512;
  v8f ar0 = v8f_zero(), ar1 = v8f_zero(), ai0 = v8f_zero(), ai1 = v8f_zero();
#pragma unroll 4
  for (int ch = 0; ch < 32; ++ch) {
    v16bf a   = *(const v16bf*)(abase + (size_t)ch * 512 + lane * 16);
    v16bf br0 = *(const v16bf*)(wr + ((size_t)ch * 4 + ct0) * 512 + lane * 16);
    v16bf br1 = *(const v16bf*)(wr + ((size_t)ch * 4 + ct0 + 1) * 512 + lane * 16);
    v16bf bi0 = *(const v16bf*)(wi + ((size_t)ch * 4 + ct0) * 512 + lane * 16);
    v16bf bi1 = *(const v16bf*)(wi + ((size_t)ch * 4 + ct0 + 1) * 512 + lane * 16);
    ar0 = wmma_bf16(a, br0, ar0);
    ar1 = wmma_bf16(a, br1, ar1);
    ai0 = wmma_bf16(a, bi0, ai0);
    ai1 = wmma_bf16(a, bi1, ai1);
  }
  int n = lane & 15, rlo = 8 * (lane >> 4);
#pragma unroll
  for (int j = 0; j < 8; ++j) {
    int s = strip * 16 + j + rlo;
    int o0 = ct0 * 16 + n, o1 = o0 + 16;
    size_t base0 = (((size_t)(b * 64 + s) * 64 + o0) * kNM + mode) * 2;
    size_t base1 = (((size_t)(b * 64 + s) * 64 + o1) * kNM + mode) * 2;
    yh[base0] = ar0[j]; yh[base0 + 1] = ai0[j];
    yh[base1] = ar1[j]; yh[base1 + 1] = ai1[j];
  }
}

// ---------------- final irfft2 (truncated 16x16 -> 32x32) ----------------
__global__ void idft_k(const float* __restrict__ yh, float* __restrict__ out) {
  __shared__ float yre[256], yim[256];
  __shared__ float gr[512], gi[512];
  int blk = blockIdx.x;                          // (b*64+s)*64+o
  int t = threadIdx.x;
  const float* src = yh + (size_t)blk * 512;
  yre[t] = src[2 * t]; yim[t] = src[2 * t + 1];
  __syncthreads();
  for (int id = t; id < 512; id += 256) {        // ifft along x (complex)
    int x = id >> 4, ky = id & 15;
    float ar = 0.f, ai = 0.f;
    for (int kx = 0; kx < 16; ++kx) {
      float sn, cs;
      __sincosf(2.0f * kPI * (float)(kx * x) / 32.0f, &sn, &cs);
      float vr = yre[kx * 16 + ky], vi = yim[kx * 16 + ky];
      ar += vr * cs - vi * sn;
      ai += vr * sn + vi * cs;
    }
    gr[id] = ar; gi[id] = ai;
  }
  __syncthreads();
  float* dst = out + (size_t)blk * 1024;
  for (int id = t; id < 1024; id += 256) {       // c2r along y (DC imag dropped)
    int x = id >> 5, y = id & 31;
    float acc = gr[x * 16];
    for (int ky = 1; ky < 16; ++ky) {
      float sn, cs;
      __sincosf(2.0f * kPI * (float)(ky * y) / 32.0f, &sn, &cs);
      acc += 2.0f * (gr[x * 16 + ky] * cs - gi[x * 16 + ky] * sn);
    }
    dst[id] = acc * (1.0f / 1024.0f);
  }
}

// ---------------- host launcher ----------------
extern "C" void kernel_launch(void* const* d_in, const int* in_sizes, int n_in,
                              void* d_out, int out_size, void* d_ws, size_t ws_size,
                              hipStream_t stream) {
  (void)in_sizes; (void)n_in; (void)out_size; (void)ws_size;
  const float* seq = (const float*)d_in[0];
  const float* wq  = (const float*)d_in[1];
  const float* wk  = (const float*)d_in[2];
  const float* wv  = (const float*)d_in[3];
  const float* wo  = (const float*)d_in[4];
  const float* cw1 = (const float*)d_in[5];
  const float* cb1 = (const float*)d_in[6];
  const float* cw2 = (const float*)d_in[7];
  const float* cb2 = (const float*)d_in[8];

  char* ws = (char*)d_ws;
  size_t off = 0;
  auto alloc = [&](size_t bytes) { size_t o = off; off += (bytes + 255) & ~(size_t)255; return o; };
  size_t o_bias  = alloc((size_t)kNH * kS * kS * 4);
  size_t o_scr   = alloc((size_t)kB * kNH * kS * kS * 4);
  size_t o_part  = alloc((size_t)8 * 16 * kS * kS * 4);
  size_t o_xa    = alloc((size_t)2 * kNM * 8 * 4 * 512 * 2);
  size_t o_wqb   = alloc((size_t)kNM * 4 * 32 * 512 * 2);
  size_t o_wkb   = alloc((size_t)kNM * 4 * 32 * 512 * 2);
  size_t o_wvb   = alloc((size_t)kNM * 4 * 32 * 512 * 2);
  size_t o_wob   = alloc((size_t)2 * kNM * 32 * 4 * 512 * 2);
  size_t o_qs    = alloc((size_t)kB * kNH * 4 * 1024 * 512 * 2);
  size_t o_kb    = alloc((size_t)kB * kNH * 4 * 1024 * 512 * 2);
  size_t o_vb    = alloc((size_t)kB * kNM * 64 * 2 * 512 * 2);
  size_t o_attn  = alloc((size_t)kB * kNH * 4 * 2 * 512 * 2);
  size_t o_oa    = alloc((size_t)kB * kNM * 4 * 32 * 512 * 2);
  size_t o_yh    = alloc((size_t)kB * kS * kC * kNM * 2 * 4);

  __bf16* xa   = (__bf16*)(ws + o_xa);
  __bf16* wqb  = (__bf16*)(ws + o_wqb);
  __bf16* wkb  = (__bf16*)(ws + o_wkb);
  __bf16* wvb  = (__bf16*)(ws + o_wvb);
  __bf16* wob  = (__bf16*)(ws + o_wob);
  __bf16* qs   = (__bf16*)(ws + o_qs);
  __bf16* kbuf = (__bf16*)(ws + o_kb);
  __bf16* vbuf = (__bf16*)(ws + o_vb);
  __bf16* attn = (__bf16*)(ws + o_attn);
  __bf16* oa   = (__bf16*)(ws + o_oa);
  float*  bias = (float*)(ws + o_bias);
  float*  scr  = (float*)(ws + o_scr);
  float*  part = (float*)(ws + o_part);
  float*  yh   = (float*)(ws + o_yh);

  prep_w_qkv<<<4096, 256, 0, stream>>>(wq, wqb);
  prep_w_qkv<<<4096, 256, 0, stream>>>(wk, wkb);
  prep_w_qkv<<<4096, 256, 0, stream>>>(wv, wvb);
  prep_w_o<<<8192, 256, 0, stream>>>(wo, wob);
  cpb_bias_k<<<16, 256, 0, stream>>>(cw1, cb1, cw2, cb2, bias);
  dft_fwd<<<8192, 256, 0, stream>>>(seq, xa);
  proj_qkv<<<2048, 256, 0, stream>>>(xa, wqb, qs, 0);
  proj_qkv<<<2048, 256, 0, stream>>>(xa, wkb, kbuf, 1);
  proj_qkv<<<2048, 256, 0, stream>>>(xa, wvb, vbuf, 2);
  score_gemm_tdm<<<128, 256, 131072, stream>>>(qs, kbuf, part);
  score_reduce<<<256, 256, 0, stream>>>(part, bias, scr);
  softmax_k<<<4, 256, 0, stream>>>(scr, attn);
  attn_v<<<kB * kNM * 4, 256, 0, stream>>>(attn, vbuf, oa);
  out_proj<<<kB * kNM, 256, 0, stream>>>(oa, wob, yh);
  idft_k<<<8192, 256, 0, stream>>>(yh, (float*)d_out);
}
